// MultiHeadAttention_1168231105139
// MI455X (gfx1250) — matrix operand order repaired and checked by reading, not compiled, not run on hardware
//
#include <hip/hip_runtime.h>
#include <hip/hip_bf16.h>

// ---------------------------------------------------------------------------
// MultiHeadAttention forward for MI455X (gfx1250), wave32 + WMMA bf16 path.
//   B=2, S=2048, E=1024, H=16, D=64
// Pipeline: f32->bf16 convert -> 3x GEMM (Q/K/V proj, scatter to [B,H,S,D])
//           -> flash attention (online softmax, WMMA for QK^T and PV)
//           -> GEMM (output proj, fp32 result + bias)
//
// All WMMA operand fetches from LDS are k-contiguous per lane so they lower
// to ds_load_b128 (the B operands are staged TRANSPOSED in LDS to make this
// true for them as well; the transpose happens on the store side, which
// needs no dependency waits).
// ---------------------------------------------------------------------------

typedef __bf16 bf16_t;
typedef __attribute__((ext_vector_type(16))) __bf16 v16bf;
typedef __attribute__((ext_vector_type(8)))  float  v8f;

#define EMBED  1024
#define SEQ    2048
#define BATCH  2
#define HEADS  16
#define HDIM   64
#define MROWS  (BATCH * SEQ)   // 4096

__device__ __forceinline__ bf16_t f2bf(float f) {
  // round-to-nearest-even fp32 -> bf16
  unsigned int u = __builtin_bit_cast(unsigned int, f);
  u += 0x7FFFu + ((u >> 16) & 1u);
  unsigned short h = (unsigned short)(u >> 16);
  return __builtin_bit_cast(bf16_t, h);
}

__device__ __forceinline__ v8f wmma_bf16(v16bf a, v16bf b, v8f c) {
  // D = A(16x32 bf16) * B(32x16 bf16) + C(16x16 f32)
  return __builtin_amdgcn_wmma_f32_16x16x32_bf16(
      /*neg_a=*/false, a, /*neg_b=*/false, b,
      /*c_mod=*/(short)0, c, /*reuse_a=*/false, /*reuse_b=*/false);
}

// A-fragment: logical A[m][k], m = lane%16; k is split as two contiguous
// 8-element chunks per lane (half*8 and 16+half*8) -> 2x ds_load_b128.
__device__ __forceinline__ v16bf load_frag_a(const bf16_t* src, int ld) {
  int lane = threadIdx.x & 31;
  int m = lane & 15, half = lane >> 4;
  const bf16_t* row = src + m * ld;
  v16bf a;
#pragma unroll
  for (int j = 0; j < 8; ++j) {
    int kb = (j < 4) ? (half * 8 + 2 * j) : (16 + half * 8 + 2 * (j - 4));
    a[2 * j]     = row[kb];
    a[2 * j + 1] = row[kb + 1];
  }
  return a;
}

// B-fragment from a TRANSPOSED tile: logical B[k][n] = src[n*ld + k].
// Per lane this reads two 8-element k chunks (half*8 and 16+half*8)
// -> 2x ds_load_b128.
__device__ __forceinline__ v16bf load_frag_bt(const bf16_t* src, int ld) {
  int lane = threadIdx.x & 31;
  int n = lane & 15, half = lane >> 4;
  const bf16_t* row = src + n * ld;
  v16bf b;
#pragma unroll
  for (int j = 0; j < 8; ++j) {
    int k = (j < 4) ? (half * 8 + 2 * j) : (16 + half * 8 + 2 * (j - 4));
    b[2 * j]     = row[k];
    b[2 * j + 1] = row[k + 1];
  }
  return b;
}

// ---------------------------------------------------------------------------
// fp32 -> bf16 elementwise convert
// ---------------------------------------------------------------------------
__global__ void convert_f32_bf16(const float* __restrict__ in,
                                 bf16_t* __restrict__ out, int n) {
  int i = blockIdx.x * blockDim.x + threadIdx.x;
  int stride = gridDim.x * blockDim.x;
  for (; i < n; i += stride) out[i] = f2bf(in[i]);
}

// ---------------------------------------------------------------------------
// Tiled bf16 GEMM: C[M,N] = A[M,K] * B[K,N] + bias[N]
//   block tile 128x128, 256 threads (8 waves), wave tile 32x64, k-step 32.
// mode 0: write fp32 row-major (final projection -> d_out)
// mode 1: write bf16 scattered to [B,H,S,D] (Q/K/V projections)
// ---------------------------------------------------------------------------
#define GA_LD 40    // 128x32 A tile rows: 80B (16B aligned, bank-clean)
#define GBT_LD 40   // transposed 128x32 B tile rows: 80B (16B aligned)

__global__ __launch_bounds__(256)
void gemm_bf16_kernel(const bf16_t* __restrict__ A, const bf16_t* __restrict__ Bm,
                      const float* __restrict__ bias, void* __restrict__ Cout,
                      int M, int N, int K, int mode) {
  __shared__ __align__(16) bf16_t As[128 * GA_LD];
  __shared__ __align__(16) bf16_t BsT[128 * GBT_LD];  // [n][k] layout

  int tid  = threadIdx.x;
  int lane = tid & 31, wave = tid >> 5;
  int wr = wave & 3;      // 0..3 -> M direction (32 rows each)
  int wc = wave >> 2;     // 0..1 -> N direction (64 cols each)
  int rowBase = blockIdx.y * 128;
  int colBase = blockIdx.x * 128;
  int half = lane >> 4, nl = lane & 15;

  v8f acc[2][4];
#pragma unroll
  for (int mt = 0; mt < 2; ++mt)
#pragma unroll
    for (int nt = 0; nt < 4; ++nt) { v8f z = {}; acc[mt][nt] = z; }

  for (int k0 = 0; k0 < K; k0 += 32) {
    // stage A tile (128x32) -> row-major LDS, b128 in / b128 out
#pragma unroll
    for (int p = 0; p < 2; ++p) {
      int flat = (p * 256 + tid) * 8;
      int r = flat >> 5, c = flat & 31;
      uint4 v = *(const uint4*)(A + (size_t)(rowBase + r) * K + k0 + c);
      *(uint4*)(&As[r * GA_LD + c]) = v;
    }
    // stage B tile (32x128) -> TRANSPOSED LDS [n][k]; scatter b16 stores
#pragma unroll
    for (int p = 0; p < 2; ++p) {
      int flat = (p * 256 + tid) * 8;
      int r = flat >> 7, c = flat & 127;          // r = k row, c = n col
      uint4 v = *(const uint4*)(Bm + (size_t)(k0 + r) * N + colBase + c);
      const unsigned short* u = (const unsigned short*)&v;
#pragma unroll
      for (int i = 0; i < 8; ++i)
        BsT[(c + i) * GBT_LD + r] = __builtin_bit_cast(bf16_t, u[i]);
    }
    // prefetch next k-step tiles into cache while this step computes
    if (k0 + 32 < K) {
      int flatA = tid * 8;
      __builtin_prefetch(A + (size_t)(rowBase + (flatA >> 5)) * K + (k0 + 32) +
                             (flatA & 31), 0, 0);
      int flatB = tid * 8;
      __builtin_prefetch(Bm + (size_t)(k0 + 32 + (flatB >> 7)) * N + colBase +
                              (flatB & 127), 0, 0);
    }
    __syncthreads();

    v16bf af[2], bfr[4];
#pragma unroll
    for (int mt = 0; mt < 2; ++mt)
      af[mt] = load_frag_a(&As[(wr * 32 + mt * 16) * GA_LD], GA_LD);
#pragma unroll
    for (int nt = 0; nt < 4; ++nt)
      bfr[nt] = load_frag_bt(&BsT[(wc * 64 + nt * 16) * GBT_LD], GBT_LD);
#pragma unroll
    for (int mt = 0; mt < 2; ++mt)
#pragma unroll
      for (int nt = 0; nt < 4; ++nt)
        acc[mt][nt] = wmma_bf16(af[mt], bfr[nt], acc[mt][nt]);
    __syncthreads();
  }

  float bcol[4];
#pragma unroll
  for (int nt = 0; nt < 4; ++nt)
    bcol[nt] = bias[colBase + wc * 64 + nt * 16 + nl];

  if (mode == 0) {
    float* Cf = (float*)Cout;
#pragma unroll
    for (int mt = 0; mt < 2; ++mt)
#pragma unroll
      for (int nt = 0; nt < 4; ++nt) {
        int c = colBase + wc * 64 + nt * 16 + nl;
#pragma unroll
        for (int v = 0; v < 8; ++v) {
          int r = rowBase + wr * 32 + mt * 16 + v + 8 * half;
          Cf[(size_t)r * N + c] = acc[mt][nt][v] + bcol[nt];
        }
      }
  } else {
    bf16_t* Cb = (bf16_t*)Cout;
#pragma unroll
    for (int mt = 0; mt < 2; ++mt)
#pragma unroll
      for (int nt = 0; nt < 4; ++nt) {
        int c = colBase + wc * 64 + nt * 16 + nl;
        int h = c >> 6, d = c & 63;
#pragma unroll
        for (int v = 0; v < 8; ++v) {
          int r = rowBase + wr * 32 + mt * 16 + v + 8 * half;
          int b = r >> 11, s = r & 2047;
          Cb[(((size_t)(b * HEADS + h) * SEQ) + s) * HDIM + d] =
              f2bf(acc[mt][nt][v] + bcol[nt]);
        }
      }
  }
}

// ---------------------------------------------------------------------------
// Flash attention: one workgroup (4 waves) per (head, 64-query tile).
// Each wave owns 16 query rows; loops over 64-key tiles with online softmax.
// Q/K/V are bf16 [B*H, S, D]; output bf16 written as [B, S, H*D] for the
// final projection GEMM.
// K tile kept natural [key][d] (QK^T needs B=K^T -> bt loader works as-is).
// V tile stored TRANSPOSED [d][key] so the PV B-fragments are contiguous.
// ---------------------------------------------------------------------------
#define KV_LD 72    // 64-row tiles padded: 144B rows (16B aligned, bank-clean)
#define P_LD  72    // per-wave 16x64 P scratch, same padding

__global__ __launch_bounds__(128)
void attention_kernel(const bf16_t* __restrict__ Q, const bf16_t* __restrict__ Kg,
                      const bf16_t* __restrict__ Vg, bf16_t* __restrict__ Og) {
  __shared__ __align__(16) bf16_t Ks[64 * KV_LD];    // [key][d]
  __shared__ __align__(16) bf16_t VsT[64 * KV_LD];   // [d][key]
  __shared__ __align__(16) bf16_t Ps[4][16 * P_LD];

  int tid  = threadIdx.x;
  int lane = tid & 31, wave = tid >> 5;
  int half = lane >> 4, nl = lane & 15;
  int bh = blockIdx.y;                 // 0..31  (= b*16 + h)
  int qbase = blockIdx.x * 64 + wave * 16;

  const bf16_t* Qh = Q  + (size_t)bh * SEQ * HDIM;
  const bf16_t* Kh = Kg + (size_t)bh * SEQ * HDIM;
  const bf16_t* Vh = Vg + (size_t)bh * SEQ * HDIM;

  // Q fragments for this wave's 16 rows (k = 0..31 and 32..63)
  v16bf qa0 = load_frag_a(Qh + (size_t)qbase * HDIM,      HDIM);
  v16bf qa1 = load_frag_a(Qh + (size_t)qbase * HDIM + 32, HDIM);

  float m_run[8], l_run[8];
  v8f O[4];
#pragma unroll
  for (int v = 0; v < 8; ++v) { m_run[v] = -1e30f; l_run[v] = 0.0f; }
#pragma unroll
  for (int dt = 0; dt < 4; ++dt) { v8f z = {}; O[dt] = z; }

  const float sc = 0.125f;  // 1/sqrt(HDIM)

  for (int kt = 0; kt < SEQ; kt += 64) {
    // cooperative load: K tile natural layout (b128 in/out), V tile
    // transposed into [d][key] via b16 scatter stores
#pragma unroll
    for (int p = 0; p < 4; ++p) {
      int flat = (p * 128 + tid) * 8;   // 0..4095
      int r = flat >> 6, c = flat & 63; // r = key row, c = d col
      *(uint4*)(&Ks[r * KV_LD + c]) =
          *(const uint4*)(Kh + (size_t)(kt + r) * HDIM + c);
      uint4 vv = *(const uint4*)(Vh + (size_t)(kt + r) * HDIM + c);
      const unsigned short* u = (const unsigned short*)&vv;
#pragma unroll
      for (int i = 0; i < 8; ++i)
        VsT[(c + i) * KV_LD + r] = __builtin_bit_cast(bf16_t, u[i]);
    }
    // prefetch next tile while this one computes
    if (kt + 64 < SEQ) {
      int flat = tid * 8;
      __builtin_prefetch(Kh + (size_t)(kt + 64 + (flat >> 6)) * HDIM +
                             (flat & 63), 0, 0);
      __builtin_prefetch(Vh + (size_t)(kt + 64 + (flat >> 6)) * HDIM +
                             (flat & 63), 0, 0);
    }
    __syncthreads();

    // S = Q * K^T  (4 key sub-tiles of 16, each 2 WMMA k-steps over D=64)
    v8f s[4];
#pragma unroll
    for (int nt = 0; nt < 4; ++nt) {
      v8f z = {};
      z = wmma_bf16(qa0, load_frag_bt(&Ks[(nt * 16) * KV_LD],      KV_LD), z);
      z = wmma_bf16(qa1, load_frag_bt(&Ks[(nt * 16) * KV_LD + 32], KV_LD), z);
      s[nt] = z;
    }

    // online softmax; rows m = v + 8*half live across the 16-lane column group
    float alpha[8], rsum[8];
#pragma unroll
    for (int v = 0; v < 8; ++v) {
      float mx = fmaxf(fmaxf(s[0][v], s[1][v]), fmaxf(s[2][v], s[3][v]));
#pragma unroll
      for (int msk = 1; msk <= 8; msk <<= 1)
        mx = fmaxf(mx, __shfl_xor(mx, msk, 32));
      float mn = fmaxf(m_run[v], mx * sc);
      alpha[v] = __expf(m_run[v] - mn);
      m_run[v] = mn;
      rsum[v] = 0.0f;
    }
#pragma unroll
    for (int nt = 0; nt < 4; ++nt) {
#pragma unroll
      for (int v = 0; v < 8; ++v) {
        float p = __expf(s[nt][v] * sc - m_run[v]);
        rsum[v] += p;
        // C-layout -> A-layout relayout through per-wave LDS scratch
        Ps[wave][(v + 8 * half) * P_LD + nt * 16 + nl] = f2bf(p);
      }
    }
#pragma unroll
    for (int v = 0; v < 8; ++v) {
      float rs = rsum[v];
#pragma unroll
      for (int msk = 1; msk <= 8; msk <<= 1) rs += __shfl_xor(rs, msk, 32);
      l_run[v] = l_run[v] * alpha[v] + rs;
    }
#pragma unroll
    for (int dt = 0; dt < 4; ++dt)
#pragma unroll
      for (int v = 0; v < 8; ++v) O[dt][v] *= alpha[v];

    // O += P * V   (LDS ops are in-order within a wave: no barrier needed
    // between the Ps stores above and these loads)
    v16bf pa0 = load_frag_a(&Ps[wave][0],  P_LD);
    v16bf pa1 = load_frag_a(&Ps[wave][32], P_LD);
#pragma unroll
    for (int dt = 0; dt < 4; ++dt) {
      O[dt] = wmma_bf16(pa0, load_frag_bt(&VsT[(dt * 16) * KV_LD],      KV_LD), O[dt]);
      O[dt] = wmma_bf16(pa1, load_frag_bt(&VsT[(dt * 16) * KV_LD + 32], KV_LD), O[dt]);
    }
    __syncthreads();
  }

  // normalize and write out as [B, S, H*D] bf16
  int b = bh >> 4, h = bh & 15;
#pragma unroll
  for (int dt = 0; dt < 4; ++dt) {
#pragma unroll
    for (int v = 0; v < 8; ++v) {
      int r = qbase + v + 8 * half;
      float val = O[dt][v] / l_run[v];
      Og[((size_t)b * SEQ + r) * EMBED + h * HDIM + dt * 16 + nl] = f2bf(val);
    }
  }
}

// ---------------------------------------------------------------------------
// host-side launch
// ---------------------------------------------------------------------------
extern "C" void kernel_launch(void* const* d_in, const int* in_sizes, int n_in,
                              void* d_out, int out_size, void* d_ws, size_t ws_size,
                              hipStream_t stream) {
  const float* x  = (const float*)d_in[0];
  const float* Wq = (const float*)d_in[1];
  const float* bq = (const float*)d_in[2];
  const float* Wk = (const float*)d_in[3];
  const float* bk = (const float*)d_in[4];
  const float* Wv = (const float*)d_in[5];
  const float* bv = (const float*)d_in[6];
  const float* Wo = (const float*)d_in[7];
  const float* bo = (const float*)d_in[8];
  float* out = (float*)d_out;

  const size_t NX = (size_t)MROWS * EMBED;               // 4M elems
  const size_t NW = (size_t)EMBED * EMBED;               // 1M elems
  const size_t NQ = (size_t)BATCH * HEADS * SEQ * HDIM;  // 4M elems

  char* ws = (char*)d_ws;
  size_t off = 0;
  bf16_t* x_bf  = (bf16_t*)(ws + off); off += NX * 2;
  bf16_t* Wq_bf = (bf16_t*)(ws + off); off += NW * 2;
  bf16_t* Wk_bf = (bf16_t*)(ws + off); off += NW * 2;
  bf16_t* Wv_bf = (bf16_t*)(ws + off); off += NW * 2;
  bf16_t* Wo_bf = (bf16_t*)(ws + off); off += NW * 2;
  bf16_t* Q_bf  = (bf16_t*)(ws + off); off += NQ * 2;
  bf16_t* K_bf  = (bf16_t*)(ws + off); off += NQ * 2;
  bf16_t* V_bf  = (bf16_t*)(ws + off); off += NQ * 2;
  bf16_t* A_bf  = (bf16_t*)(ws + off); off += NX * 2;   // attention output
  if (off > ws_size) return;  // workspace too small: nothing sane to do

  convert_f32_bf16<<<2048, 256, 0, stream>>>(x,  x_bf,  (int)NX);
  convert_f32_bf16<<<1024, 256, 0, stream>>>(Wq, Wq_bf, (int)NW);
  convert_f32_bf16<<<1024, 256, 0, stream>>>(Wk, Wk_bf, (int)NW);
  convert_f32_bf16<<<1024, 256, 0, stream>>>(Wv, Wv_bf, (int)NW);
  convert_f32_bf16<<<1024, 256, 0, stream>>>(Wo, Wo_bf, (int)NW);

  dim3 ggrid(EMBED / 128, MROWS / 128);  // (8, 32)
  gemm_bf16_kernel<<<ggrid, 256, 0, stream>>>(x_bf, Wq_bf, bq, Q_bf,
                                              MROWS, EMBED, EMBED, 1);
  gemm_bf16_kernel<<<ggrid, 256, 0, stream>>>(x_bf, Wk_bf, bk, K_bf,
                                              MROWS, EMBED, EMBED, 1);
  gemm_bf16_kernel<<<ggrid, 256, 0, stream>>>(x_bf, Wv_bf, bv, V_bf,
                                              MROWS, EMBED, EMBED, 1);

  dim3 agrid(SEQ / 64, BATCH * HEADS);   // (32, 32)
  attention_kernel<<<agrid, 128, 0, stream>>>(Q_bf, K_bf, V_bf, A_bf);

  gemm_bf16_kernel<<<ggrid, 256, 0, stream>>>(A_bf, Wo_bf, bo, out,
                                              MROWS, EMBED, EMBED, 0);
}
